// Model_37778532335921
// MI455X (gfx1250) — matrix operand order repaired and checked by reading, not compiled, not run on hardware
//
#include <hip/hip_runtime.h>
#include <math.h>

// ---------------------------------------------------------------------------
// CDNA5 / gfx1250 implementation. wave32. Matrix ops use
// v_wmma_f32_16x16x32_bf16 (f32 accumulate, bf16 inputs); tile staging uses
// double-buffered gfx1250 async global->LDS copies (ASYNCcnt), one barrier
// per K-slab.
// ---------------------------------------------------------------------------

typedef __attribute__((ext_vector_type(16))) __bf16       v16bf;
typedef __attribute__((ext_vector_type(8)))  float        v8f;
typedef __attribute__((ext_vector_type(4)))  unsigned int u32x4;
typedef int i32x4 __attribute__((vector_size(16)));       // matches builtin proto

#if __has_builtin(__builtin_amdgcn_global_load_async_to_lds_b128) && \
    __has_builtin(__builtin_amdgcn_s_wait_asynccnt)
#define TSF_ASYNC 1
typedef __attribute__((address_space(1))) i32x4 tsf_gvec;  // global int4
typedef __attribute__((address_space(3))) i32x4 tsf_lvec;  // LDS int4

__device__ __forceinline__ void tsf_async_copy16(const void* g, void* l)
{
    __builtin_amdgcn_global_load_async_to_lds_b128(
        (tsf_gvec*)(i32x4*)(void*)g, (tsf_lvec*)(i32x4*)l, 0, 0);
}
#else
#define TSF_ASYNC 0
#endif

// ---------------- model constants (from reference) -------------------------
#define BSZ   16
#define TSEQ  512
#define CIN   32
#define NN1   512
#define DST   16
#define RNK   32
#define PREDL 96

// ---------------- GEMM tiling ----------------------------------------------
// Block tile 64(M) x 128(N) x 32(K); 8 wave32s in a 2x4 grid; each wave owns a
// 32x32 C tile = 4 WMMA accumulators. Requires K % 32 == 0 and 16B-aligned
// operands (host guarantees both). Double-buffered LDS staging.
#define BM  64
#define BN  128
#define BKK 32
#define LDK 40   // padded LDS K-stride (bf16 elems) => 80B row stride (16B aligned)

// act: 0 none, 1 relu, 2 gelu(exact), 3 silu, 4 softplus
__global__ __launch_bounds__(256)
void tsf_gemm(const __bf16* __restrict__ A,   // [M,K] row-major, 16B aligned
              const __bf16* __restrict__ Wt,  // [N,K] row-major (linear weight)
              const float*  __restrict__ bias,// [N] or nullptr
              float* __restrict__ C,          // [M,N]
              int M, int N, int K, int act)
{
    __shared__ __bf16 As[2][BM * LDK];
    __shared__ __bf16 Bs[2][BN * LDK];

    const int tid  = threadIdx.x;
    const int lane = tid & 31;
    const int wv   = tid >> 5;        // 0..7
    const int wm   = wv >> 2;         // 0..1 : 32-row group of C
    const int wn   = wv & 3;          // 0..3 : 32-col group of C
    const int hf   = lane >> 4;       // lane half
    const int l16  = lane & 15;

    const int m0 = blockIdx.y * BM;
    const int n0 = blockIdx.x * BN;

    v8f acc[2][2];
#pragma unroll
    for (int s = 0; s < 2; ++s)
#pragma unroll
        for (int c = 0; c < 2; ++c) acc[s][c] = (v8f){};

    // staging: thread moves one 16B chunk of A, two 16B chunks of B per slab
    const int srow = tid >> 2;            // 0..63
    const int scol = (tid & 3) * 8;       // 0,8,16,24
    const int agr  = (m0 + srow < M) ? (m0 + srow) : (M - 1);          // clamp
    const int bgr0 = (n0 + srow < N) ? (n0 + srow) : (N - 1);
    const int bgr1 = (n0 + srow + 64 < N) ? (n0 + srow + 64) : (N - 1);

    const __bf16* agp  = A  + (size_t)agr  * K + scol;   // slab-0 chunk bases
    const __bf16* bgp0 = Wt + (size_t)bgr0 * K + scol;
    const __bf16* bgp1 = Wt + (size_t)bgr1 * K + scol;
    const int loff  = srow * LDK + scol;                 // LDS chunk offsets
    const int loff1 = (srow + 64) * LDK + scol;

    const int nsteps = K / BKK;

#if TSF_ASYNC
    // prologue: slab 0 -> buffer 0
    tsf_async_copy16(agp,  &As[0][loff]);
    tsf_async_copy16(bgp0, &Bs[0][loff]);
    tsf_async_copy16(bgp1, &Bs[0][loff1]);
#endif

    for (int k = 0; k < nsteps; ++k) {
        const int cb = k & 1;
#if TSF_ASYNC
        __builtin_amdgcn_s_wait_asynccnt(0);   // own slab-k copies landed
        __syncthreads();                       // everyone's copies visible;
                                               // everyone done with slab k-1
        if (k + 1 < nsteps) {                  // overlap slab k+1 with compute
            const int nbuf = (k + 1) & 1;
            const int kc   = (k + 1) * BKK;
            tsf_async_copy16(agp  + kc, &As[nbuf][loff]);
            tsf_async_copy16(bgp0 + kc, &Bs[nbuf][loff]);
            tsf_async_copy16(bgp1 + kc, &Bs[nbuf][loff1]);
        }
        if (k + 2 < nsteps) {                  // keep slab k+2 flowing into L2
            const int kp = (k + 2) * BKK;
            __builtin_prefetch(agp  + kp, 0, 1);
            __builtin_prefetch(bgp0 + kp, 0, 1);
            __builtin_prefetch(bgp1 + kp, 0, 1);
        }
#else
        const int kc = k * BKK;
        *(u32x4*)&As[0][loff]  = *(const u32x4*)(agp  + kc);
        *(u32x4*)&Bs[0][loff]  = *(const u32x4*)(bgp0 + kc);
        *(u32x4*)&Bs[0][loff1] = *(const u32x4*)(bgp1 + kc);
        __syncthreads();
#endif

        // fragment gathers (ISA VGPR layouts, contiguous 32-bit pairs)
        v16bf afr[2], bfr[2];
#pragma unroll
        for (int s = 0; s < 2; ++s) {
#if TSF_ASYNC
            const __bf16* arow = &As[cb][(wm * 32 + s * 16 + l16) * LDK];
            const __bf16* brow = &Bs[cb][(wn * 32 + s * 16 + l16) * LDK];
#else
            const __bf16* arow = &As[0][(wm * 32 + s * 16 + l16) * LDK];
            const __bf16* brow = &Bs[0][(wn * 32 + s * 16 + l16) * LDK];
#endif
#pragma unroll
            for (int j = 0; j < 8; ++j) {
                const int kk = (j < 4) ? (hf * 8 + 2 * j)
                                       : (16 + hf * 8 + 2 * (j - 4));
                afr[s][2 * j]     = arow[kk];
                afr[s][2 * j + 1] = arow[kk + 1];
            }
#pragma unroll
            for (int j = 0; j < 8; ++j) {
                const int kk = (j < 4) ? (hf * 8 + 2 * j)
                                       : (16 + hf * 8 + 2 * (j - 4));
                bfr[s][2 * j]     = brow[kk];
                bfr[s][2 * j + 1] = brow[kk + 1];
            }
        }
#pragma unroll
        for (int s = 0; s < 2; ++s)
#pragma unroll
            for (int c = 0; c < 2; ++c)
                acc[s][c] = __builtin_amdgcn_wmma_f32_16x16x32_bf16(
                    false, afr[s], false, bfr[c], (short)0, acc[s][c],
                    false, false);
#if !TSF_ASYNC
        __syncthreads();
#endif
        (void)cb;
    }

    // epilogue: C layout VGPR r -> M = r + 8*hf, N = lane&15
#pragma unroll
    for (int s = 0; s < 2; ++s) {
#pragma unroll
        for (int c = 0; c < 2; ++c) {
#pragma unroll
            for (int r = 0; r < 8; ++r) {
                const int m = m0 + wm * 32 + s * 16 + r + hf * 8;
                const int n = n0 + wn * 32 + c * 16 + l16;
                if (m < M && n < N) {
                    float v = acc[s][c][r];
                    if (bias) v += bias[n];
                    if (act == 1)      v = fmaxf(v, 0.f);
                    else if (act == 2) v = 0.5f * v * (1.f + erff(v * 0.70710678f));
                    else if (act == 3) v = v / (1.f + __expf(-v));
                    else if (act == 4) v = (v > 20.f) ? v : log1pf(__expf(v));
                    C[(size_t)m * N + n] = v;
                }
            }
        }
    }
}

// ---------------- elementwise / small kernels ------------------------------

__global__ void tsf_cvt_bf16(const float* __restrict__ s, __bf16* __restrict__ d, int n)
{
    int i = blockIdx.x * blockDim.x + threadIdx.x;
    if (i < n) d[i] = (__bf16)s[i];
}

// per-(b,c) instance norm over T; also records mean/std for the output head
__global__ void tsf_instnorm(const float* __restrict__ x, float* __restrict__ xn,
                             float* __restrict__ mean, float* __restrict__ stdev,
                             int Bn, int L, int Cc)
{
    int idx = blockIdx.x * blockDim.x + threadIdx.x;
    if (idx >= Bn * Cc) return;
    int c = idx % Cc, b = idx / Cc;
    float s = 0.f, s2 = 0.f;
    for (int l = 0; l < L; ++l) {
        float v = x[((size_t)b * L + l) * Cc + c];
        s += v; s2 += v * v;
    }
    float m  = s / L;
    float sd = sqrtf(fmaxf(s2 / L - m * m, 0.f) + 1e-5f);
    mean[idx] = m; stdev[idx] = sd;
    float inv = 1.f / sd;
    for (int l = 0; l < L; ++l) {
        size_t o = ((size_t)b * L + l) * Cc + c;
        xn[o] = (x[o] - m) * inv;
    }
}

// circular conv (k=3, C->D) token embedding + sinusoidal PE
__global__ void tsf_embed(const float* __restrict__ x,  // [B,L,C]
                          const float* __restrict__ w,  // [D,C,3]
                          float* __restrict__ y,        // [B,L,D]
                          int Bn, int L, int Cc, int D)
{
    int idx = blockIdx.x * blockDim.x + threadIdx.x;
    if (idx >= Bn * L * D) return;
    int d = idx % D, l = (idx / D) % L, b = idx / (D * L);
    float s = 0.f;
#pragma unroll
    for (int k = 0; k < 3; ++k) {
        int ls = l + k - 1; ls = (ls + L) % L;
        const float* xr = x + ((size_t)b * L + ls) * Cc;
        const float* wr = w + (size_t)d * Cc * 3 + k;
        for (int c = 0; c < Cc; ++c) s += xr[c] * wr[c * 3];
    }
    int   i2  = d >> 1;
    float div = __expf(-logf(10000.f) * (2.f * (float)i2) / (float)D);
    float ang = (float)l * div;
    s += (d & 1) ? __cosf(ang) : __sinf(ang);
    y[idx] = s;
}

// causal depthwise conv + bias + silu
__global__ void tsf_dwconv_silu(const float* __restrict__ x, const float* __restrict__ w,
                                const float* __restrict__ bv, float* __restrict__ y,
                                int Bn, int L, int D, int K)
{
    int idx = blockIdx.x * blockDim.x + threadIdx.x;
    if (idx >= Bn * L * D) return;
    int d = idx % D, l = (idx / D) % L, b = idx / (D * L);
    float s = bv[d];
    for (int k = 0; k < K; ++k) {
        int ls = l - (K - 1) + k;
        if (ls >= 0) s += x[((size_t)b * L + ls) * D + d] * w[d * K + k];
    }
    y[idx] = s / (1.f + __expf(-s));
}

// selective scan; one thread per (b,d); NS states in registers.
// cond==1: delta is [B,D] and Bv/Cv are [B,NS] (STIM-style conditioning).
template<int NS>
__global__ void tsf_selscan(const float* __restrict__ u, const float* __restrict__ delta,
                            const float* __restrict__ A_log, const float* __restrict__ Bv,
                            const float* __restrict__ Cv, const float* __restrict__ Dv,
                            float* __restrict__ y, int Bn, int L, int D, int cond)
{
    int idx = blockIdx.x * blockDim.x + threadIdx.x;
    if (idx >= Bn * D) return;
    int d = idx % D, b = idx / D;
    float Ar[NS], h[NS];
#pragma unroll
    for (int n = 0; n < NS; ++n) { Ar[n] = -__expf(A_log[(size_t)d * NS + n]); h[n] = 0.f; }
    const float Dd = Dv[d];
    for (int l = 0; l < L; ++l) {
        const size_t bi = (size_t)b * L + l;
        const float  ut = u[bi * D + d];
        const float  dt = cond ? delta[(size_t)b * D + d] : delta[bi * D + d];
        const float* Br = cond ? (Bv + (size_t)b * NS) : (Bv + bi * NS);
        const float* Cr = cond ? (Cv + (size_t)b * NS) : (Cv + bi * NS);
        const float  du = dt * ut;
        float out = 0.f;
#pragma unroll
        for (int n = 0; n < NS; ++n) {
            h[n] = __expf(dt * Ar[n]) * h[n] + du * Br[n];
            out += h[n] * Cr[n];
        }
        y[bi * D + d] = out + ut * Dd;
    }
}

__global__ void tsf_mul_silu(const float* __restrict__ a, const float* __restrict__ z,
                             float* __restrict__ y, int n)
{
    int i = blockIdx.x * blockDim.x + threadIdx.x;
    if (i < n) { float zv = z[i]; y[i] = a[i] * (zv / (1.f + __expf(-zv))); }
}

__global__ void tsf_add3(const float* __restrict__ a, const float* __restrict__ b,
                         const float* __restrict__ c, float* __restrict__ y, int n)
{
    int i = blockIdx.x * blockDim.x + threadIdx.x;
    if (i < n) y[i] = a[i] + b[i] + c[i];
}

// y = LN(x (+res)) * g + b ; one 256-thread block per row
__global__ void tsf_layernorm(const float* __restrict__ x, const float* __restrict__ res,
                              const float* __restrict__ g, const float* __restrict__ bt,
                              float* __restrict__ y, int D)
{
    const int row = blockIdx.x, tid = threadIdx.x;
    __shared__ float s1[8], s2[8], smv[2];
    const float* xr = x + (size_t)row * D;
    const float* rr = res ? res + (size_t)row * D : nullptr;
    float ls = 0.f, lq = 0.f;
    for (int i = tid; i < D; i += 256) {
        float v = xr[i] + (rr ? rr[i] : 0.f);
        ls += v; lq += v * v;
    }
    for (int o = 16; o > 0; o >>= 1) { ls += __shfl_xor(ls, o, 32); lq += __shfl_xor(lq, o, 32); }
    if ((tid & 31) == 0) { s1[tid >> 5] = ls; s2[tid >> 5] = lq; }
    __syncthreads();
    if (tid == 0) {
        float a = 0.f, q = 0.f;
        for (int i = 0; i < 8; ++i) { a += s1[i]; q += s2[i]; }
        smv[0] = a / D; smv[1] = q / D;
    }
    __syncthreads();
    const float mean = smv[0];
    const float inv  = rsqrtf(fmaxf(smv[1] - mean * mean, 0.f) + 1e-5f);
    for (int i = tid; i < D; i += 256) {
        float v = xr[i] + (rr ? rr[i] : 0.f);
        y[(size_t)row * D + i] = (v - mean) * inv * g[i] + bt[i];
    }
}

__global__ void tsf_flipL(const float* __restrict__ s, float* __restrict__ d,
                          int Bn, int L, int D)
{
    int idx = blockIdx.x * blockDim.x + threadIdx.x;
    if (idx >= Bn * L * D) return;
    int dd = idx % D, l = (idx / D) % L, b = idx / (D * L);
    d[((size_t)b * L + (L - 1 - l)) * D + dd] = s[idx];
}

__global__ void tsf_copy_cols(const float* __restrict__ s, float* __restrict__ d,
                              int M, int N, int c0, int nc)
{
    int idx = blockIdx.x * blockDim.x + threadIdx.x;
    if (idx >= M * nc) return;
    int c = idx % nc, m = idx / nc;
    d[idx] = s[(size_t)m * N + c0 + c];
}

__global__ void tsf_gather_rows(const float* __restrict__ s, float* __restrict__ d,
                                int Bn, int T, int Cc, int D)
{
    int idx = blockIdx.x * blockDim.x + threadIdx.x;
    if (idx >= Bn * Cc * D) return;
    int dd = idx % D, c = (idx / D) % Cc, b = idx / (D * Cc);
    d[((size_t)(b * Cc + c)) * D + dd] = s[((size_t)b * T + c) * D + dd];
}

// out[b,p,c] = t[(b*C+c)*P + p] * stdev[b,c] + mean[b,c]
__global__ void tsf_out_final(const float* __restrict__ t, const float* __restrict__ mean,
                              const float* __restrict__ stdev, float* __restrict__ o,
                              int Bn, int Cc, int P)
{
    int idx = blockIdx.x * blockDim.x + threadIdx.x;
    if (idx >= Bn * P * Cc) return;
    int c = idx % Cc, p = (idx / Cc) % P, b = idx / (Cc * P);
    float v = t[((size_t)(b * Cc + c)) * P + p];
    o[idx] = v * stdev[b * Cc + c] + mean[b * Cc + c];
}

// ---------------------------------------------------------------------------
// host orchestration
// ---------------------------------------------------------------------------

extern "C" void kernel_launch(void* const* d_in, const int* in_sizes, int n_in,
                              void* d_out, int out_size, void* d_ws, size_t ws_size,
                              hipStream_t stream)
{
    (void)out_size; (void)ws_size;
    const int  Bn = BSZ, T = TSEQ, Cc = CIN, D = NN1;
    const int  M  = Bn * T;                       // 8192 token rows

    // ---- locate x vs params (jax pytree flatten sorts dict keys => params
    //      first, x last; but be robust to insertion order too) ------------
    const float* xin;
    int pi;
    if (n_in > 0 && in_sizes[0] == Bn * T * Cc) { xin = (const float*)d_in[0]; pi = 1; }
    else { xin = (const float*)d_in[n_in - 1]; pi = 0; }

    auto nf = [&]() -> const float* {
        int i = (pi < n_in) ? pi : (n_in - 1);
        ++pi;
        return (const float*)d_in[i];
    };

    struct MW { const float *A_log, *Dv, *conv_b, *conv_w, *dt_b, *dt_w, *in_w, *out_w, *xproj_w; };
    auto readM = [&]() { MW m; m.A_log = nf(); m.Dv = nf(); m.conv_b = nf(); m.conv_w = nf();
                         m.dt_b = nf(); m.dt_w = nf(); m.in_w = nf(); m.out_w = nf();
                         m.xproj_w = nf(); return m; };
    struct LW { const float *c1_b, *c1_w, *c2_b, *c2_w, *ln1_b, *ln1_g, *ln2_b, *ln2_g; MW m1, m2; };
    auto readL = [&]() { LW l; l.c1_b = nf(); l.c1_w = nf(); l.c2_b = nf(); l.c2_w = nf();
                         l.ln1_b = nf(); l.ln1_g = nf(); l.ln2_b = nf(); l.ln2_g = nf();
                         l.m1 = readM(); l.m2 = readM(); return l; };

    const float* emb1_w = nf();
    nf();                                        // emb2_w (graph path: later round)
    LW eL[2]; eL[0] = readL(); eL[1] = readL();
    const float* e_ln_b = nf();
    const float* e_ln_g = nf();
    for (int i = 0; i < 54; ++i) nf();           // enc2
    for (int i = 0; i < 12; ++i) nf();           // g
    for (int i = 0; i < 12; ++i) nf();           // inc1 + inc2
    nf(); nf();                                  // l1_b, l1_w
    const float* l2_b = nf(); const float* l2_w = nf();
    const float* l3_b = nf(); const float* l3_w = nf();
    nf(); nf(); nf(); nf();                      // lcat_b, lcat_w, n2n1_b, n2n1_w
    for (int i = 0; i < 11; ++i) nf();           // stim

    // ---- workspace bump allocator (256B-aligned blocks) -------------------
    char* wsp = (char*)d_ws;
    auto bumpB = [&](size_t bytes) -> void* {
        void* r = wsp; wsp += (bytes + 255) & ~(size_t)255; return r;
    };
    const size_t rowf = (size_t)M * D * sizeof(float);
    float*  xn    = (float*)bumpB((size_t)M * Cc * sizeof(float));
    float*  mean  = (float*)bumpB((size_t)Bn * Cc * sizeof(float));
    float*  stdev = (float*)bumpB((size_t)Bn * Cc * sizeof(float));
    float*  xcur  = (float*)bumpB(rowf);
    float*  y1    = (float*)bumpB(rowf);
    float*  y2    = (float*)bumpB(rowf);
    float*  xflip = (float*)bumpB(rowf);
    float*  t0    = (float*)bumpB(rowf);
    float*  t1    = (float*)bumpB(rowf);
    float*  t2    = (float*)bumpB(rowf);
    float*  t3    = (float*)bumpB(rowf);
    float*  t4    = (float*)bumpB(rowf);
    float*  xz    = (float*)bumpB(rowf * 2);
    float*  dbc   = (float*)bumpB((size_t)M * (RNK + 2 * DST) * sizeof(float));
    float*  dtb   = (float*)bumpB((size_t)M * RNK * sizeof(float));
    float*  Bsl   = (float*)bumpB((size_t)M * DST * sizeof(float));
    float*  Csl   = (float*)bumpB((size_t)M * DST * sizeof(float));
    float*  fout  = (float*)bumpB((size_t)Bn * Cc * PREDL * sizeof(float));
    __bf16* abuf  = (__bf16*)bumpB((size_t)M * D * sizeof(__bf16));
    __bf16* wbuf  = (__bf16*)bumpB((size_t)2 * D * D * sizeof(__bf16));

    auto nb = [](int n) { return (n + 255) / 256; };
    auto cvt = [&](const float* s, __bf16* d, int n) {
        tsf_cvt_bf16<<<nb(n), 256, 0, stream>>>(s, d, n);
    };
    auto gemm = [&](const __bf16* A, const __bf16* W, const float* bias, float* C,
                    int Mm, int Nn, int Kk, int act) {
        dim3 g((Nn + BN - 1) / BN, (Mm + BM - 1) / BM);
        tsf_gemm<<<g, 256, 0, stream>>>(A, W, bias, C, Mm, Nn, Kk, act);
    };
    auto lnorm = [&](const float* x, const float* res, const float* g,
                     const float* b, float* y, int rows, int Dd) {
        tsf_layernorm<<<rows, 256, 0, stream>>>(x, res, g, b, y, Dd);
    };

    // ---- one Mamba block: xin_[M,D] -> yout[M,D] --------------------------
    auto mamba = [&](const float* xsrc, const MW& p, float* yout) {
        cvt(xsrc, abuf, M * D);
        cvt(p.in_w, wbuf, 2 * D * D);
        gemm(abuf, wbuf, nullptr, xz, M, 2 * D, D, 0);                 // in-proj
        tsf_copy_cols<<<nb(M * D), 256, 0, stream>>>(xz, t0, M, 2 * D, 0, D);   // xi
        tsf_copy_cols<<<nb(M * D), 256, 0, stream>>>(xz, t1, M, 2 * D, D, D);   // z
        tsf_dwconv_silu<<<nb(M * D), 256, 0, stream>>>(t0, p.conv_w, p.conv_b,
                                                       t2, Bn, T, D, 2);
        cvt(t2, abuf, M * D);
        cvt(p.xproj_w, wbuf, (RNK + 2 * DST) * D);
        gemm(abuf, wbuf, nullptr, dbc, M, RNK + 2 * DST, D, 0);        // x-proj
        tsf_copy_cols<<<nb(M * RNK), 256, 0, stream>>>(dbc, dtb, M, RNK + 2 * DST, 0, RNK);
        tsf_copy_cols<<<nb(M * DST), 256, 0, stream>>>(dbc, Bsl, M, RNK + 2 * DST, RNK, DST);
        tsf_copy_cols<<<nb(M * DST), 256, 0, stream>>>(dbc, Csl, M, RNK + 2 * DST, RNK + DST, DST);
        cvt(dtb, abuf, M * RNK);
        cvt(p.dt_w, wbuf, D * RNK);
        gemm(abuf, wbuf, p.dt_b, t3, M, D, RNK, 4);                    // delta = softplus
        tsf_selscan<DST><<<nb(Bn * D), 256, 0, stream>>>(t2, t3, p.A_log, Bsl, Csl,
                                                         p.Dv, t4, Bn, T, D, 0);
        tsf_mul_silu<<<nb(M * D), 256, 0, stream>>>(t4, t1, t3, M * D); // y * silu(z)
        cvt(t3, abuf, M * D);
        cvt(p.out_w, wbuf, D * D);
        gemm(abuf, wbuf, nullptr, yout, M, D, D, 0);                   // out-proj
    };

    // ---- forward: instance norm -> embed -> encoder -> head ---------------
    tsf_instnorm<<<nb(Bn * Cc), 256, 0, stream>>>(xin, xn, mean, stdev, Bn, T, Cc);
    tsf_embed<<<nb(M * D), 256, 0, stream>>>(xn, emb1_w, xcur, Bn, T, Cc, D);

    for (int l = 0; l < 2; ++l) {
        const LW& L = eL[l];
        mamba(xcur, L.m1, y1);                                         // forward pass
        tsf_flipL<<<nb(M * D), 256, 0, stream>>>(xcur, xflip, Bn, T, D);
        mamba(xflip, L.m2, y2);                                        // backward pass
        tsf_flipL<<<nb(M * D), 256, 0, stream>>>(y2, t0, Bn, T, D);
        tsf_add3<<<nb(M * D), 256, 0, stream>>>(xcur, y1, t0, xcur, M * D);
        lnorm(xcur, nullptr, L.ln1_g, L.ln1_b, xcur, M, D);
        cvt(xcur, abuf, M * D);
        cvt(L.c1_w, wbuf, D * D);
        gemm(abuf, wbuf, L.c1_b, t0, M, D, D, 1);                      // FFN relu
        cvt(t0, abuf, M * D);
        cvt(L.c2_w, wbuf, D * D);
        gemm(abuf, wbuf, L.c2_b, t1, M, D, D, 0);
        lnorm(xcur, t1, L.ln2_g, L.ln2_b, xcur, M, D);
    }
    lnorm(xcur, nullptr, e_ln_g, e_ln_b, xcur, M, D);

    // ---- head: per-channel rows -> l2 (D->T) -> l3 (T->PRED) -> denorm ----
    const int Mh = Bn * Cc;                                            // 512 rows
    tsf_gather_rows<<<nb(Mh * D), 256, 0, stream>>>(xcur, t0, Bn, T, Cc, D);
    cvt(t0, abuf, Mh * D);
    cvt(l2_w, wbuf, T * D);
    gemm(abuf, wbuf, l2_b, t1, Mh, T, D, 0);
    cvt(t1, abuf, Mh * T);
    cvt(l3_w, wbuf, PREDL * T);
    gemm(abuf, wbuf, l3_b, fout, Mh, PREDL, T, 0);
    tsf_out_final<<<nb(Bn * PREDL * Cc), 256, 0, stream>>>(fout, mean, stdev,
                                                           (float*)d_out, Bn, Cc, PREDL);
}
